// EdgeModelDiff_5909875000172
// MI455X (gfx1250) — matrix operand order repaired and checked by reading, not compiled, not run on hardware
//
#include <hip/hip_runtime.h>

typedef __attribute__((ext_vector_type(16))) _Float16 v16h;
typedef __attribute__((ext_vector_type(8)))  _Float16 v8h;
typedef __attribute__((ext_vector_type(8)))  float    v8f;
typedef __attribute__((ext_vector_type(4)))  float    v4f;

#define XSTRIDE 80          // halves per LDS row (160 B: 16B-aligned, 2-way bank spread)
#define WPB 8               // waves per block (256 threads, wave32)
#define DIN 48              // real K of GEMM1 (padded to 64 with zeros)

__device__ __forceinline__ void wave_ds_sync() {
    // same-wave LDS RAW/WAR fence: DS ops are in-order per wave; this is a
    // compiler barrier + hardware wait on DScnt.
    asm volatile("s_wait_dscnt 0" ::: "memory");
}

__global__ void __launch_bounds__(256, 1)
edge_model_wmma(const float* __restrict__ src,
                const float* __restrict__ dest,
                const float* __restrict__ edge_attr,
                const float* __restrict__ u,
                const int*   __restrict__ batch,
                const float* __restrict__ W1,
                const float* __restrict__ b1,
                const float* __restrict__ W2,
                const float* __restrict__ b2,
                float* __restrict__ out,
                int E)
{
    __shared__ __align__(16) _Float16 Xs_all[WPB * 16 * XSTRIDE];
    __shared__ __align__(16) _Float16 Hs_all[WPB * 16 * XSTRIDE];

    const int lane = threadIdx.x & 31;
    const int wave = threadIdx.x >> 5;
    const int nl   = lane & 15;   // M index (A) / N index (B,C/D)
    const int hi   = lane >> 4;   // half-wave selector

    _Float16* Xs = &Xs_all[wave * 16 * XSTRIDE];
    _Float16* Hs = &Hs_all[wave * 16 * XSTRIDE];

    // ---------------- persistent weight B-fragments (built once) ----------------
    // B 32x16 f16 layout: lane n = l&15, element i -> K = 32c + (l>>4)*16 + i
    v16h w1f[4][2];
#pragma unroll
    for (int t = 0; t < 4; ++t)
#pragma unroll
        for (int c = 0; c < 2; ++c)
#pragma unroll
            for (int i = 0; i < 16; ++i) {
                int K = 32 * c + hi * 16 + i;
                float w = (K < DIN) ? W1[K * 64 + 16 * t + nl] : 0.0f;
                w1f[t][c][i] = (_Float16)w;
            }
    v16h w2f[2];
#pragma unroll
    for (int c = 0; c < 2; ++c)
#pragma unroll
        for (int i = 0; i < 16; ++i) {
            int K = 32 * c + hi * 16 + i;
            w2f[c][i] = (_Float16)W2[K * 16 + nl];
        }
    float b1v[4];
#pragma unroll
    for (int t = 0; t < 4; ++t) b1v[t] = b1[16 * t + nl];
    const float b2v = b2[nl];

    // zero-pad X columns 48..63 once (never overwritten afterwards)
    {
        v8h z = {};
        *(v8h*)&Xs[(lane >> 1) * XSTRIDE + DIN + 8 * (lane & 1)] = z;
    }

    const int ntiles = (E + 15) >> 4;
    const int tstep  = gridDim.x * WPB;

    for (int tile = blockIdx.x * WPB + wave; tile < ntiles; tile += tstep) {
        const int base = tile << 4;

        // prefetch next tile's edge rows (global_prefetch_b8)
        {
            int en = base + tstep * 16 + (lane >> 1);
            if (en < E) {
                __builtin_prefetch(&dest[en * 16], 0, 1);
                __builtin_prefetch(&src[en * 16], 0, 1);
                __builtin_prefetch(&edge_attr[en * 16], 0, 1);
            }
        }

        // ---------------- stage X tile (16 edges x 48 feats) into LDS as f16 ----
        {
            int er = lane >> 1;                 // edge row 0..15
            int e  = base + er; if (e >= E) e = E - 1;
            int cg = (lane & 1) * 8;            // 8-feature column group
            v4f d0 = *(const v4f*)&dest[e * 16 + cg];
            v4f d1 = *(const v4f*)&dest[e * 16 + cg + 4];
            v4f s0 = *(const v4f*)&src [e * 16 + cg];
            v4f s1 = *(const v4f*)&src [e * 16 + cg + 4];
            v4f a0 = *(const v4f*)&edge_attr[e * 16 + cg];
            v4f a1 = *(const v4f*)&edge_attr[e * 16 + cg + 4];
            int bi = batch[e];
            v4f u0 = *(const v4f*)&u[bi * 16 + cg];
            v4f u1 = *(const v4f*)&u[bi * 16 + cg + 4];
            v8h xd, xa, xu;
#pragma unroll
            for (int j = 0; j < 4; ++j) {
                xd[j]     = (_Float16)(d0[j] - s0[j]);
                xd[4 + j] = (_Float16)(d1[j] - s1[j]);
                xa[j]     = (_Float16)a0[j];
                xa[4 + j] = (_Float16)a1[j];
                xu[j]     = (_Float16)u0[j];
                xu[4 + j] = (_Float16)u1[j];
            }
            *(v8h*)&Xs[er * XSTRIDE +      cg] = xd;   // cols 0..15 : dest-src
            *(v8h*)&Xs[er * XSTRIDE + 16 + cg] = xa;   // cols 16..31: edge_attr
            *(v8h*)&Xs[er * XSTRIDE + 32 + cg] = xu;   // cols 32..47: u[batch]
        }
        wave_ds_sync();

        // ---------------- GEMM1: H = relu(X @ W1 + b1), X is A 16x32 per chunk --
        // A f16 16x32: lane m = l&15; elems 0..15 -> K = 32c+hi*16+0..15
        //              (two contiguous b128 spans, matching B)
        v16h xa0, xa1;
        {
            v8h l0 = *(const v8h*)&Xs[nl * XSTRIDE +  0 + hi * 16];
            v8h h0 = *(const v8h*)&Xs[nl * XSTRIDE +  8 + hi * 16];
            v8h l1 = *(const v8h*)&Xs[nl * XSTRIDE + 32 + hi * 16];
            v8h h1 = *(const v8h*)&Xs[nl * XSTRIDE + 40 + hi * 16];
#pragma unroll
            for (int j = 0; j < 8; ++j) {
                xa0[j] = l0[j]; xa0[8 + j] = h0[j];
                xa1[j] = l1[j]; xa1[8 + j] = h1[j];
            }
        }
#pragma unroll
        for (int t = 0; t < 4; ++t) {
            v8f acc = {};
            acc = __builtin_amdgcn_wmma_f32_16x16x32_f16(false, xa0, false, w1f[t][0],
                                                         (short)0, acc, false, false);
            acc = __builtin_amdgcn_wmma_f32_16x16x32_f16(false, xa1, false, w1f[t][1],
                                                         (short)0, acc, false, false);
#pragma unroll
            for (int r = 0; r < 8; ++r) {        // C/D: row r+hi*8, col 16t+nl
                float h = acc[r] + b1v[t];
                h = h > 0.0f ? h : 0.0f;
                Hs[(r + hi * 8) * XSTRIDE + 16 * t + nl] = (_Float16)h;
            }
        }
        wave_ds_sync();

        // ---------------- GEMM2: out = H @ W2 + b2 ------------------------------
        v16h ha0, ha1;
        {
            v8h l0 = *(const v8h*)&Hs[nl * XSTRIDE +  0 + hi * 16];
            v8h h0 = *(const v8h*)&Hs[nl * XSTRIDE +  8 + hi * 16];
            v8h l1 = *(const v8h*)&Hs[nl * XSTRIDE + 32 + hi * 16];
            v8h h1 = *(const v8h*)&Hs[nl * XSTRIDE + 40 + hi * 16];
#pragma unroll
            for (int j = 0; j < 8; ++j) {
                ha0[j] = l0[j]; ha0[8 + j] = h0[j];
                ha1[j] = l1[j]; ha1[8 + j] = h1[j];
            }
        }
        v8f acc2 = {};
        acc2 = __builtin_amdgcn_wmma_f32_16x16x32_f16(false, ha0, false, w2f[0],
                                                      (short)0, acc2, false, false);
        acc2 = __builtin_amdgcn_wmma_f32_16x16x32_f16(false, ha1, false, w2f[1],
                                                      (short)0, acc2, false, false);
#pragma unroll
        for (int r = 0; r < 8; ++r) {
            int e = base + r + hi * 8;
            if (e < E) out[e * 16 + nl] = acc2[r] + b2v;  // each half-wave: 64B row
        }
        wave_ds_sync();   // WAR: gathers done before next iteration overwrites LDS
    }
}

extern "C" void kernel_launch(void* const* d_in, const int* in_sizes, int n_in,
                              void* d_out, int out_size, void* d_ws, size_t ws_size,
                              hipStream_t stream) {
    const float* src  = (const float*)d_in[0];
    const float* dest = (const float*)d_in[1];
    const float* ea   = (const float*)d_in[2];
    const float* u    = (const float*)d_in[3];
    const int*   bat  = (const int*)d_in[4];
    const float* W1   = (const float*)d_in[5];
    const float* b1   = (const float*)d_in[6];
    const float* W2   = (const float*)d_in[7];
    const float* b2   = (const float*)d_in[8];
    float* out = (float*)d_out;

    int E = in_sizes[0] / 16;            // src is [E,16]
    int ntiles = (E + 15) / 16;
    int nblocks = (ntiles + WPB - 1) / WPB;
    if (nblocks > 1024) nblocks = 1024;  // grid-stride: ~12 tiles/wave amortizes
                                         // the one-time weight-fragment build
    hipLaunchKernelGGL(edge_model_wmma, dim3(nblocks), dim3(256), 0, stream,
                       src, dest, ea, u, bat, W1, b1, W2, b2, out, E);
}
